// CustomMoEBranch_9586367004756
// MI455X (gfx1250) — matrix operand order repaired and checked by reading, not compiled, not run on hardware
//
#include <hip/hip_runtime.h>
#include <math.h>

// ---------------- problem constants ----------------
#define BATCH   256
#define LEN     4096
#define NEXP    8
#define DOUT    1024
#define HGATE   128
#define NFFT    256
#define NBIN    129   // NFFT/2 + 1
#define NFRM    33
#define HOP     128

typedef __attribute__((ext_vector_type(16))) __bf16 v16bf;
typedef __attribute__((ext_vector_type(8)))  float  v8f;

// =====================================================================
// Kernel 1: STFT-mag gating.  One block per sample, 256 threads.
// =====================================================================
__global__ void gate_kernel(const float* __restrict__ x,
                            const float* __restrict__ w1,
                            const float* __restrict__ b1,
                            const float* __restrict__ w2,
                            const float* __restrict__ b2,
                            float* __restrict__ gwt,
                            int*   __restrict__ gidx)
{
    __shared__ float sx[LEN];
    __shared__ float ctab[NFFT];
    __shared__ float stab[NFFT];
    __shared__ float fr[NFFT];
    __shared__ float sPooled[NBIN];
    __shared__ float sH[HGATE];
    __shared__ float sLog[NEXP];

    const int b   = blockIdx.x;
    const int tid = threadIdx.x;

    for (int i = tid; i < LEN; i += 256) sx[i] = x[(size_t)b * LEN + i];

    const float TWO_PI = 6.283185307179586f;
    float ang = (TWO_PI / (float)NFFT) * (float)tid;
    ctab[tid] = cosf(ang);
    stab[tid] = sinf(ang);
    __syncthreads();

    float pooled = 0.0f;
    for (int t = 0; t < NFRM; t++) {
        // windowed frame with reflect padding (pad = 128)
        {
            int i = t * HOP + tid;          // 0 .. 4351
            int j = i - 128;
            if (j < 0) j = -j;
            else if (j >= LEN) j = 2 * LEN - 2 - j;
            float win = 0.5f * (1.0f - ctab[tid]);
            fr[tid] = win * sx[j];
        }
        __syncthreads();
        if (tid < NBIN) {
            float re = 0.0f, im = 0.0f;
            int idx = 0;
            #pragma unroll 8
            for (int n = 0; n < NFFT; n++) {
                float v = fr[n];
                int p = idx & (NFFT - 1);
                re += v * ctab[p];
                im -= v * stab[p];
                idx += tid;
            }
            pooled += sqrtf(re * re + im * im);
        }
        __syncthreads();
    }
    if (tid < NBIN) sPooled[tid] = pooled * (1.0f / (float)NFRM);
    __syncthreads();

    if (tid < HGATE) {
        float h = b1[tid];
        #pragma unroll 4
        for (int f = 0; f < NBIN; f++) h += sPooled[f] * w1[f * HGATE + tid];
        sH[tid] = fmaxf(h, 0.0f);
    }
    __syncthreads();

    if (tid < NEXP) {
        float lg = b2[tid];
        #pragma unroll 8
        for (int j = 0; j < HGATE; j++) lg += sH[j] * w2[j * NEXP + tid];
        sLog[tid] = lg;
    }
    __syncthreads();

    if (tid == 0) {
        int i0 = 0; float v0 = sLog[0];
        for (int i = 1; i < NEXP; i++) if (sLog[i] > v0) { v0 = sLog[i]; i0 = i; }
        int i1 = -1; float v1 = -3.0e38f;
        for (int i = 0; i < NEXP; i++) {
            if (i == i0) continue;
            if (sLog[i] > v1) { v1 = sLog[i]; i1 = i; }
        }
        float e1 = __expf(v1 - v0);
        float s  = 1.0f + e1;
        gwt[b * 2 + 0] = 1.0f / s;
        gwt[b * 2 + 1] = e1 / s;
        gidx[b * 2 + 0] = i0;
        gidx[b * 2 + 1] = i1;
    }
}

// =====================================================================
// Kernel 2: expert GEMM, bf16 WMMA, double-buffered LDS, fused gather.
//   C[e] = X (256x4096) * W[e] (4096x1024)
// Block tile: 128(M) x 64(N) x 64(K-step); 256 threads = 8 wave32s.
// Wave grid 4x2; each wave owns 32x32 -> 4 v8f accumulators,
// 8 WMMAs per K-step, ONE barrier per K-step (ping-pong buffers).
// =====================================================================
#define BLK_M 128
#define BLK_N 64
#define BLK_K 64
#define NKT   (LEN / BLK_K)   // 64 iterations

__global__ __launch_bounds__(256)
void moe_wmma_gemm(const float* __restrict__ x,
                   const float* __restrict__ ew,
                   const float* __restrict__ eb,
                   const float* __restrict__ gwt,
                   const int*   __restrict__ gidx,
                   float* __restrict__ out)
{
    __shared__ __bf16 sA[2][BLK_M * BLK_K];   // [m][k] row-major      (16KB each)
    __shared__ __bf16 sBt[2][BLK_N * BLK_K];  // K-transposed [n][k]   (8KB each)

    const int tid   = threadIdx.x;
    const int lane  = tid & 31;
    const int wid   = tid >> 5;          // 0..7 (wave32)
    const int wm    = wid >> 1;          // 0..3 -> 32-row band
    const int wn    = wid & 1;           // 0..1 -> 32-col band
    const int lrow  = lane & 15;
    const int lhalf = lane >> 4;

    const int n0 = blockIdx.x * BLK_N;
    const int m0 = blockIdx.y * BLK_M;
    const int e  = blockIdx.z;

    const float* We = ew + (size_t)e * LEN * DOUT;

    // staging geometry (fixed per thread)
    const int ar = tid >> 1;             // A row 0..127
    const int ac = (tid & 1) * 32;       // A col 0 or 32
    const int bn = tid & 63;             // B column 0..63
    const int bk = (tid >> 6) * 16;      // B k-run start 0/16/32/48

    v8f acc[2][2] = {};

    // ---- stage tile kk into buffer buf ----
    auto stage = [&](int kk, int buf) {
        const int k0 = kk * BLK_K;
        // A: 32 contiguous f32 per thread -> 32 contiguous bf16 in LDS
        {
            const float* gx = x + (size_t)(m0 + ar) * LEN + k0 + ac;
            __bf16* sa = &sA[buf][ar * BLK_K + ac];
            #pragma unroll
            for (int j = 0; j < 32; j++) sa[j] = (__bf16)gx[j];
        }
        // B: one column, 16 k's: lane-coalesced b32 loads, contiguous LDS run
        {
            const float* gw_ = We + (size_t)(k0 + bk) * DOUT + n0 + bn;
            __bf16* sb = &sBt[buf][bn * BLK_K + bk];
            #pragma unroll
            for (int j = 0; j < 16; j++) sb[j] = (__bf16)gw_[(size_t)j * DOUT];
        }
    };

    stage(0, 0);
    __syncthreads();

    int buf = 0;
    for (int kk = 0; kk < NKT; kk++) {
        if (kk + 1 < NKT) stage(kk + 1, buf ^ 1);

        // ---- fragments: contiguous K order, same in A and B ----
        // A (16x32 bf16): lane-half selects contiguous K half
        // B (32x16 bf16): lane-half selects contiguous K half
        #pragma unroll
        for (int ks = 0; ks < 2; ks++) {
            const int kb = ks * 32;
            v16bf afr[2], bfr[2];
            #pragma unroll
            for (int rc = 0; rc < 2; rc++) {
                const __bf16* p = &sA[buf][(wm * 32 + rc * 16 + lrow) * BLK_K + kb];
                const int o1 = lhalf * 16;
                #pragma unroll
                for (int i = 0; i < 16; i++) {
                    afr[rc][i]     = p[o1 + i];
                }
            }
            #pragma unroll
            for (int cc = 0; cc < 2; cc++) {
                const __bf16* p = &sBt[buf][(wn * 32 + cc * 16 + lrow) * BLK_K + kb + lhalf * 16];
                #pragma unroll
                for (int i = 0; i < 16; i++) bfr[cc][i] = p[i];
            }
            #pragma unroll
            for (int rc = 0; rc < 2; rc++)
                #pragma unroll
                for (int cc = 0; cc < 2; cc++)
                    acc[rc][cc] = __builtin_amdgcn_wmma_f32_16x16x32_bf16(
                        false, afr[rc], false, bfr[cc],
                        (short)0, acc[rc][cc], false, false);
        }

        __syncthreads();
        buf ^= 1;
    }

    // ---- fused epilogue: bias + top-2 gather + weighted atomic add ----
    #pragma unroll
    for (int cc = 0; cc < 2; cc++) {
        const int col = n0 + wn * 32 + cc * 16 + lrow;
        const float bias = eb[e * DOUT + col];
        #pragma unroll
        for (int rc = 0; rc < 2; rc++) {
            #pragma unroll
            for (int r = 0; r < 8; r++) {
                const int brow = m0 + wm * 32 + rc * 16 + r + lhalf * 8;
                const int e0 = gidx[brow * 2 + 0];
                const int e1 = gidx[brow * 2 + 1];
                float w = 0.0f;
                if (e0 == e)      w = gwt[brow * 2 + 0];
                else if (e1 == e) w = gwt[brow * 2 + 1];
                if (w != 0.0f) {
                    float v = w * (acc[rc][cc][r] + bias);
                    __hip_atomic_fetch_add(&out[(size_t)brow * DOUT + col], v,
                                           __ATOMIC_RELAXED, __HIP_MEMORY_SCOPE_AGENT);
                }
            }
        }
    }
}

// =====================================================================
// launch
// =====================================================================
extern "C" void kernel_launch(void* const* d_in, const int* in_sizes, int n_in,
                              void* d_out, int out_size, void* d_ws, size_t ws_size,
                              hipStream_t stream)
{
    const float* x  = (const float*)d_in[0];
    const float* w1 = (const float*)d_in[1];
    const float* b1 = (const float*)d_in[2];
    const float* w2 = (const float*)d_in[3];
    const float* b2 = (const float*)d_in[4];
    const float* ew = (const float*)d_in[5];
    const float* eb = (const float*)d_in[6];
    float* out = (float*)d_out;

    float* gwt = (float*)d_ws;                 // 2*B floats
    int*   gidx = (int*)((char*)d_ws + 2048);  // 2*B ints

    hipMemsetAsync(d_out, 0, (size_t)out_size * sizeof(float), stream);

    gate_kernel<<<BATCH, 256, 0, stream>>>(x, w1, b1, w2, b2, gwt, gidx);

    dim3 grid(DOUT / BLK_N, BATCH / BLK_M, NEXP);   // 16 x 2 x 8 = 256 blocks
    moe_wmma_gemm<<<grid, 256, 0, stream>>>(x, ew, eb, gwt, gidx, out);
}
